// MPNNLSTM_9723805958403
// MI455X (gfx1250) — matrix operand order repaired and checked by reading, not compiled, not run on hardware
//
#include <hip/hip_runtime.h>

// ---------------------------------------------------------------------------
// MPNN-LSTM for MI455X (gfx1250, wave32).
//   T=3, N=50000, FIN=4, HID=128, E=1.6M
// Dense GEMMs use v_wmma_f32_16x16x32_f16 (f16 operands, f32 accum), with the
// shared A tile staged through LDS (ds_load_b128) once per 256-thread block.
// Message passing is the bandwidth-bound part: gather + f32 atomic scatter,
// with the whole working set resident in the 192MB L2.
// ---------------------------------------------------------------------------

typedef _Float16 f16;
typedef __attribute__((ext_vector_type(8)))  _Float16 v8h;
typedef __attribute__((ext_vector_type(16))) _Float16 v16h;
typedef __attribute__((ext_vector_type(8)))  float    v8f;

#define HID   128
#define TSTEPS 3
#define FINC  4
#define BN_EPS 1e-5f

static __device__ __forceinline__ float sigmf(float x) {
    return 1.0f / (1.0f + __expf(-x));
}

// ---------------------------------------------------------------- utilities
__global__ void k_fill_f32(float* __restrict__ p, float v, int n) {
    int i = blockIdx.x * blockDim.x + threadIdx.x;
    if (i < n) p[i] = v;
}
__global__ void k_fill_f16(f16* __restrict__ p, int n) {
    int i = blockIdx.x * blockDim.x + threadIdx.x;
    if (i < n) p[i] = (f16)0.0f;
}
__global__ void k_cvt_f16(const float* __restrict__ in, f16* __restrict__ out, int n) {
    int i = blockIdx.x * blockDim.x + threadIdx.x;
    if (i < n) out[i] = (f16)in[i];
}
// wc[i] is [K=128, N=128] row-major; emit wcT16[i][n*128 + k] (column's K contiguous)
__global__ void k_cvt_wcT(const float* __restrict__ wc, f16* __restrict__ out, int total) {
    int i = blockIdx.x * blockDim.x + threadIdx.x;
    if (i >= total) return;
    int l   = i / (HID * HID);
    int rem = i - l * HID * HID;
    int k   = rem / HID;
    int n   = rem - k * HID;
    out[l * HID * HID + n * HID + k] = (f16)wc[i];
}

// ------------------------------------------------------------- graph norm
__global__ void k_deg(const long long* __restrict__ ei, const float* __restrict__ w,
                      float* __restrict__ deg, int E) {
    int e = blockIdx.x * blockDim.x + threadIdx.x;
    if (e < E) atomicAdd(&deg[(int)ei[(long long)E + e]], w[e]);
}
__global__ void k_dinv(const float* __restrict__ deg, float* __restrict__ dinv, int n) {
    int i = blockIdx.x * blockDim.x + threadIdx.x;
    if (i < n) {
        float d = deg[i];
        dinv[i] = (d > 0.0f) ? rsqrtf(d) : 0.0f;
    }
}
__global__ void k_norm(const long long* __restrict__ ei, const float* __restrict__ w,
                       const float* __restrict__ dinv, float* __restrict__ nrm, int E) {
    int e = blockIdx.x * blockDim.x + threadIdx.x;
    if (e < E)
        nrm[e] = dinv[(int)ei[e]] * w[e] * dinv[(int)ei[(long long)E + e]];
}

// ------------------------------------------------- GCN layer 0 (K=4, VALU)
__global__ void k_gcn0(const float* __restrict__ X, const float* __restrict__ w1,
                       float* __restrict__ XW, int N) {
    int i = blockIdx.x * blockDim.x + threadIdx.x;
    if (i >= N * HID) return;
    int n = i >> 7, j = i & (HID - 1);
    const float* x = X + n * FINC;
    float acc = 0.0f;
#pragma unroll
    for (int f = 0; f < FINC; ++f) acc += x[f] * w1[f * HID + j];
    XW[i] = acc;
}

// ------------------------------------------------- WMMA fragment helpers
// A fragment (16x32 f16): lane<16 -> K 0..15 of its row, lane>=16 -> K 16..31
static __device__ __forceinline__ v16h load_a_frag(const f16* __restrict__ row,
                                                   int ktile, int laneHi) {
    int k0 = ktile * 32 + laneHi * 16;
    v8h lo = *(const v8h*)(row + k0);
    v8h hi = *(const v8h*)(row + k0 + 8);
    v16h f;
#pragma unroll
    for (int i = 0; i < 8; ++i) { f[i] = lo[i]; f[i + 8] = hi[i]; }
    return f;
}
// B fragment (32x16 f16): lane<16 -> K 0..15 of its column, lane>=16 -> K 16..31
static __device__ __forceinline__ v16h load_b_frag(const f16* __restrict__ col,
                                                   int ktile, int laneHi) {
    int k0 = ktile * 32 + laneHi * 16;
    v8h lo = *(const v8h*)(col + k0);
    v8h hi = *(const v8h*)(col + k0 + 8);
    v16h f;
#pragma unroll
    for (int i = 0; i < 8; ++i) { f[i] = lo[i]; f[i + 8] = hi[i]; }
    return f;
}

// Stage one 16xHID f16 tile into LDS: 256 threads x 16B each (one pass).
static __device__ __forceinline__ void stage_tile(f16* __restrict__ dst,
                                                  const f16* __restrict__ src,
                                                  int row0, int N, int tid) {
    int r  = tid >> 4;        // 0..15 row within tile
    int ch = tid & 15;        // 0..15 16-byte chunk within row
    int gr = row0 + r; if (gr > N - 1) gr = N - 1;
    *(v8h*)(dst + r * HID + ch * 8) =
        *(const v8h*)(src + (long long)gr * HID + ch * 8);
}

// --------------------------------------- GEMM: C[N,128] = A16[N,128] @ B
// BT is [128 cols][128 K] row-major f16 (each output column's K contiguous).
// Block = 256 threads = 8 waves; wave w -> col tile w; block -> 16-row stripe.
// A tile staged via LDS and shared by all 8 waves; two accumulators for ILP.
__global__ void k_gemm_f16(const f16* __restrict__ A, const f16* __restrict__ BT,
                           float* __restrict__ C, int N) {
    __shared__ f16 sA[16 * HID];
    const int tid  = threadIdx.x;
    const int row0 = blockIdx.x * 16;
    stage_tile(sA, A, row0, N, tid);
    __syncthreads();

    const int lane   = tid & 31;
    const int wave   = tid >> 5;
    const int laneHi = lane >> 4;
    const int laneLo = lane & 15;
    const int col0   = wave * 16;

    const f16* arow = sA + laneLo * HID;                 // LDS
    const f16* bcol = BT + (col0 + laneLo) * HID;        // global (L2-resident)

    v8f acc0 = {}, acc1 = {};
    acc0 = __builtin_amdgcn_wmma_f32_16x16x32_f16(false, load_a_frag(arow, 0, laneHi),
                                                  false, load_b_frag(bcol, 0, laneHi),
                                                  (short)0, acc0, false, false);
    acc1 = __builtin_amdgcn_wmma_f32_16x16x32_f16(false, load_a_frag(arow, 1, laneHi),
                                                  false, load_b_frag(bcol, 1, laneHi),
                                                  (short)0, acc1, false, false);
    acc0 = __builtin_amdgcn_wmma_f32_16x16x32_f16(false, load_a_frag(arow, 2, laneHi),
                                                  false, load_b_frag(bcol, 2, laneHi),
                                                  (short)0, acc0, false, false);
    acc1 = __builtin_amdgcn_wmma_f32_16x16x32_f16(false, load_a_frag(arow, 3, laneHi),
                                                  false, load_b_frag(bcol, 3, laneHi),
                                                  (short)0, acc1, false, false);
    v8f acc = acc0 + acc1;

    if (row0 + 16 <= N) {   // uniform fast path: straight-line coalesced stores
        float* cb = C + (long long)(row0 + laneHi * 8) * HID + col0 + laneLo;
#pragma unroll
        for (int r = 0; r < 8; ++r) cb[(long long)r * HID] = acc[r];
    } else {
#pragma unroll
        for (int r = 0; r < 8; ++r) {
            int m = row0 + r + laneHi * 8;
            if (m < N) C[(long long)m * HID + col0 + laneLo] = acc[r];
        }
    }
}

// ------------------------------------------------- scatter (message passing)
__global__ void k_scatter_init(const float* __restrict__ XW, const float* __restrict__ dinv,
                               const float* __restrict__ bias, float* __restrict__ AGG, int N) {
    int i = blockIdx.x * blockDim.x + threadIdx.x;
    if (i >= N * HID) return;
    int n = i >> 7, j = i & (HID - 1);
    float dv = dinv[n];
    AGG[i] = XW[i] * dv * dv + bias[j];   // self loop (norm = dinv^2) + bias
}
__global__ void k_scatter_edges(const long long* __restrict__ ei,
                                const float* __restrict__ nrm,
                                const float* __restrict__ XW,
                                float* __restrict__ AGG, int E) {
    int t = blockIdx.x * blockDim.x + threadIdx.x;
    int e = t >> 5;
    if (e >= E) return;
    int lane = t & 31;
    int s = (int)ei[e];
    int d = (int)ei[(long long)E + e];
    float nm = nrm[e];
    float4 v = *(const float4*)(XW + (long long)s * HID + lane * 4);
    float* dst = AGG + (long long)d * HID + lane * 4;
    atomicAdd(dst + 0, nm * v.x);
    atomicAdd(dst + 1, nm * v.y);
    atomicAdd(dst + 2, nm * v.z);
    atomicAdd(dst + 3, nm * v.w);
}

// ------------------------------------------------- BatchNorm over relu(AGG)
__global__ void k_bn_stats(const float* __restrict__ AGG, float* __restrict__ sum,
                           float* __restrict__ sumsq, int N) {
    int c  = threadIdx.x;                 // 128 threads
    int n0 = blockIdx.x * 64;
    int n1 = n0 + 64; if (n1 > N) n1 = N;
    float s = 0.0f, s2 = 0.0f;
    for (int n = n0; n < n1; ++n) {
        float v = AGG[(long long)n * HID + c];
        v = v > 0.0f ? v : 0.0f;          // relu before BN stats
        s += v; s2 += v * v;
    }
    atomicAdd(&sum[c], s);
    atomicAdd(&sumsq[c], s2);
}
__global__ void k_bn_final(const float* __restrict__ sum, const float* __restrict__ sumsq,
                           float* __restrict__ mu, float* __restrict__ rstd, float invN) {
    int c = threadIdx.x;
    float m = sum[c] * invN;
    float v = sumsq[c] * invN - m * m;
    mu[c] = m;
    rstd[c] = rsqrtf(v + BN_EPS);
}
__global__ void k_bn_apply(const float* __restrict__ AGG, const float* __restrict__ mu,
                           const float* __restrict__ rstd, const float* __restrict__ g,
                           const float* __restrict__ beta, f16* __restrict__ dst16, int N) {
    int i = blockIdx.x * blockDim.x + threadIdx.x;
    if (i >= N * HID) return;
    int c = i & (HID - 1);
    float v = AGG[i];
    v = v > 0.0f ? v : 0.0f;
    float y = (v - mu[c]) * rstd[c] * g[c] + beta[c];
    dst16[i] = (f16)y;
}

// ------------------------------------------------- fused LSTM step (WMMA)
// Each wave: 16 rows x one 16-col tile of i/f/g/o gates (4 accumulators),
// gates = x@wih.T + h@whh.T + b, then activations + c/h update in epilogue.
// x/h tiles staged via LDS; wih16/whh16 are [512 rows][128 K] row-major f16.
__global__ void k_lstm_step(const f16* __restrict__ x16, const f16* __restrict__ h16,
                            const f16* __restrict__ wih16, const f16* __restrict__ whh16,
                            const float* __restrict__ b, float* __restrict__ c32,
                            float* __restrict__ h32, f16* __restrict__ h16out,
                            f16* __restrict__ seq16out, int N) {
    __shared__ f16 sX[16 * HID];
    __shared__ f16 sH[16 * HID];
    const int tid  = threadIdx.x;
    const int row0 = blockIdx.x * 16;
    stage_tile(sX, x16, row0, N, tid);
    stage_tile(sH, h16, row0, N, tid);
    __syncthreads();

    const int lane   = tid & 31;
    const int wave   = tid >> 5;
    const int laneHi = lane >> 4;
    const int laneLo = lane & 15;
    const int col0   = wave * 16;

    const f16* xrow = sX + laneLo * HID;
    const f16* hrow = sH + laneLo * HID;

    v8f acc[4] = {{}, {}, {}, {}};
#pragma unroll
    for (int kt = 0; kt < 4; ++kt) {
        v16h ax = load_a_frag(xrow, kt, laneHi);
        v16h ah = load_a_frag(hrow, kt, laneHi);
#pragma unroll
        for (int g = 0; g < 4; ++g) {
            const f16* bi = wih16 + (g * HID + col0 + laneLo) * HID;
            const f16* bh = whh16 + (g * HID + col0 + laneLo) * HID;
            v16h bfi = load_b_frag(bi, kt, laneHi);
            acc[g] = __builtin_amdgcn_wmma_f32_16x16x32_f16(false, ax, false, bfi,
                                                            (short)0, acc[g], false, false);
            v16h bfh = load_b_frag(bh, kt, laneHi);
            acc[g] = __builtin_amdgcn_wmma_f32_16x16x32_f16(false, ah, false, bfh,
                                                            (short)0, acc[g], false, false);
        }
    }

    const int col  = col0 + laneLo;
    const bool full = (row0 + 16 <= N);
#pragma unroll
    for (int r = 0; r < 8; ++r) {
        int m = row0 + r + laneHi * 8;
        if (!full && m >= N) continue;
        long long idx = (long long)m * HID + col;
        float iv = sigmf(acc[0][r] + b[col]);
        float fv = sigmf(acc[1][r] + b[HID + col]);
        float gv = tanhf(acc[2][r] + b[2 * HID + col]);
        float ov = sigmf(acc[3][r] + b[3 * HID + col]);
        float cn = fv * c32[idx] + iv * gv;
        float hn = ov * tanhf(cn);
        c32[idx] = cn;
        h32[idx] = hn;
        f16 hh = (f16)hn;
        h16out[idx] = hh;
        if (seq16out) seq16out[idx] = hh;
    }
}

// ------------------------------------------------- regression head
__global__ void k_head1(const float* __restrict__ h32, const float* __restrict__ X,
                        const float* __restrict__ l1w, const float* __restrict__ l1b,
                        float* __restrict__ tmp, int N) {
    int n = blockIdx.x;              // one block per node, 128 threads
    int j = threadIdx.x;
    if (n >= N) return;
    const float* hr = h32 + (long long)n * HID;
    float acc = l1b[j];
    for (int k = 0; k < HID; ++k) {
        float v = hr[k];
        v = v > 0.0f ? v : 0.0f;     // relu(hT)
        acc += v * l1w[k * HID + j];
    }
#pragma unroll
    for (int t = 0; t < TSTEPS; ++t) {
        float s = X[((long long)t * N + n) * FINC];   // X[t, n, 0]
        acc += s * l1w[(HID + t) * HID + j];
    }
    tmp[(long long)n * HID + j] = acc;
}
__global__ void k_head2(const float* __restrict__ tmp, const float* __restrict__ l2w,
                        const float* __restrict__ l2b, float* __restrict__ out, int N) {
    int n = blockIdx.x * blockDim.x + threadIdx.x;
    if (n >= N) return;
    const float* r = tmp + (long long)n * HID;
    float acc = l2b[0];
    for (int k = 0; k < HID; ++k) acc += r[k] * l2w[k];
    out[n] = acc;
}

// ---------------------------------------------------------------------------
extern "C" void kernel_launch(void* const* d_in, const int* in_sizes, int n_in,
                              void* d_out, int out_size, void* d_ws, size_t ws_size,
                              hipStream_t stream) {
    const float*     X    = (const float*)d_in[0];
    const long long* EI   = (const long long*)d_in[1];
    const float*     EW   = (const float*)d_in[2];
    const float*     W1   = (const float*)d_in[3];
    const float*     WC   = (const float*)d_in[4];
    const float*     BC   = (const float*)d_in[5];
    const float*     G    = (const float*)d_in[6];
    const float*     BETA = (const float*)d_in[7];
    const float*     WIH  = (const float*)d_in[8];
    const float*     WHH  = (const float*)d_in[9];
    const float*     LB   = (const float*)d_in[10];
    const float*     L1W  = (const float*)d_in[11];
    const float*     L1B  = (const float*)d_in[12];
    const float*     L2W  = (const float*)d_in[13];
    const float*     L2B  = (const float*)d_in[14];

    const int N  = in_sizes[0] / (TSTEPS * FINC);
    const int E  = in_sizes[2];
    const long long NH = (long long)N * HID;

    // ---- carve workspace --------------------------------------------------
    char* base = (char*)d_ws;
    size_t off = 0;
    auto carve = [&](size_t bytes) -> char* {
        char* p = base + off;
        off = (off + bytes + 255) & ~(size_t)255;
        return p;
    };
    float* deg    = (float*)carve((size_t)N * 4);
    float* dinv   = (float*)carve((size_t)N * 4);
    float* nrm    = (float*)carve((size_t)E * 4);
    float* XW     = (float*)carve(NH * 4);           // GEMM out / head tmp
    float* AGG    = (float*)carve(NH * 4);           // scatter accumulator
    float* bnS    = (float*)carve(128 * 4);
    float* bnS2   = (float*)carve(128 * 4);
    float* bnMu   = (float*)carve(128 * 4);
    float* bnRs   = (float*)carve(128 * 4);
    f16*   Hcur16 = (f16*)carve(NH * 2);             // f16 activations (GCN chain)
    f16*   Cseq16 = (f16*)carve(3 * NH * 2);         // per-timestep GCN outputs
    f16*   SeqB16 = (f16*)carve(3 * NH * 2);         // LSTM layer-0 output seq
    f16*   hA16   = (f16*)carve(NH * 2);
    f16*   hB16   = (f16*)carve(NH * 2);
    float* h32    = (float*)carve(NH * 4);
    float* c32    = (float*)carve(NH * 4);
    f16*   wcT16  = (f16*)carve((size_t)3 * HID * HID * 2);
    f16*   wih16  = (f16*)carve((size_t)2 * 4 * HID * HID * 2);
    f16*   whh16  = (f16*)carve((size_t)2 * 4 * HID * HID * 2);
    (void)ws_size;

    const int TPB = 256;
    auto blk = [&](long long n) { return (int)((n + TPB - 1) / TPB); };
    const int gRows = (N + 15) / 16;                 // WMMA row-tile grid

    // ---- graph normalization ---------------------------------------------
    k_fill_f32<<<blk(N), TPB, 0, stream>>>(deg, 1.0f, N);   // self-loop weight
    k_deg<<<blk(E), TPB, 0, stream>>>(EI, EW, deg, E);
    k_dinv<<<blk(N), TPB, 0, stream>>>(deg, dinv, N);
    k_norm<<<blk(E), TPB, 0, stream>>>(EI, EW, dinv, nrm, E);

    // ---- weight conversion to f16 (B layout: column K contiguous) --------
    k_cvt_wcT<<<blk(3 * HID * HID), TPB, 0, stream>>>(WC, wcT16, 3 * HID * HID);
    k_cvt_f16<<<blk(2 * 4 * HID * HID), TPB, 0, stream>>>(WIH, wih16, 2 * 4 * HID * HID);
    k_cvt_f16<<<blk(2 * 4 * HID * HID), TPB, 0, stream>>>(WHH, whh16, 2 * 4 * HID * HID);

    // ---- GCN stack per timestep ------------------------------------------
    auto gcn_post = [&](int layer, f16* dst16) {   // scatter + relu + BN -> f16
        k_scatter_init<<<blk(NH), TPB, 0, stream>>>(XW, dinv, BC + layer * HID, AGG, N);
        k_scatter_edges<<<blk((long long)E * 32), TPB, 0, stream>>>(EI, nrm, XW, AGG, E);
        k_fill_f32<<<1, 256, 0, stream>>>(bnS, 0.0f, 128);
        k_fill_f32<<<1, 256, 0, stream>>>(bnS2, 0.0f, 128);
        k_bn_stats<<<(N + 63) / 64, 128, 0, stream>>>(AGG, bnS, bnS2, N);
        k_bn_final<<<1, 128, 0, stream>>>(bnS, bnS2, bnMu, bnRs, 1.0f / (float)N);
        k_bn_apply<<<blk(NH), TPB, 0, stream>>>(AGG, bnMu, bnRs, G + layer * HID,
                                                BETA + layer * HID, dst16, N);
    };

    for (int t = 0; t < TSTEPS; ++t) {
        // layer 0: K=4 transform (VALU), then scatter+BN
        k_gcn0<<<blk(NH), TPB, 0, stream>>>(X + (long long)t * N * FINC, W1, XW, N);
        gcn_post(0, Hcur16);
        // layers 1..3: WMMA transform on f16 activations
        for (int l = 0; l < 3; ++l) {
            k_gemm_f16<<<gRows, TPB, 0, stream>>>(Hcur16, wcT16 + l * HID * HID, XW, N);
            f16* dst = (l < 2) ? Hcur16 : (Cseq16 + (long long)t * NH);
            gcn_post(l + 1, dst);
        }
    }

    // ---- 2-layer LSTM over T=3 -------------------------------------------
    for (int layer = 0; layer < 2; ++layer) {
        k_fill_f16<<<blk(NH), TPB, 0, stream>>>(hA16, (int)NH);
        k_fill_f32<<<blk(NH), TPB, 0, stream>>>(c32, 0.0f, (int)NH);
        const f16* seqIn  = layer ? SeqB16 : Cseq16;
        f16*       seqOut = layer ? (f16*)nullptr : SeqB16;
        f16* hin = hA16;
        f16* hout = hB16;
        for (int t = 0; t < TSTEPS; ++t) {
            k_lstm_step<<<gRows, TPB, 0, stream>>>(
                seqIn + (long long)t * NH, hin,
                wih16 + (size_t)layer * 4 * HID * HID,
                whh16 + (size_t)layer * 4 * HID * HID,
                LB + layer * 4 * HID, c32, h32, hout,
                seqOut ? seqOut + (long long)t * NH : (f16*)nullptr, N);
            f16* tmp = hin; hin = hout; hout = tmp;
        }
    }

    // ---- head -------------------------------------------------------------
    k_head1<<<N, 128, 0, stream>>>(h32, X, L1W, L1B, XW, N);
    k_head2<<<blk(N), TPB, 0, stream>>>(XW, L2W, L2B, (float*)d_out, N);
}
